// MLP_APPNP_5755256177438
// MI455X (gfx1250) — matrix operand order repaired and checked by reading, not compiled, not run on hardware
//
#include <hip/hip_runtime.h>

#define IN_DIM 256
#define HIDDEN 64
#define ODIM   2
#define KSTEPS 10
#define ALPHA  0.1f

typedef __attribute__((ext_vector_type(16))) __bf16 v16bf;
typedef __attribute__((ext_vector_type(8)))  float  v8f;

__device__ __forceinline__ void pack8(v16bf& d, int base, float4 lo, float4 hi) {
    d[base+0] = (__bf16)lo.x; d[base+1] = (__bf16)lo.y;
    d[base+2] = (__bf16)lo.z; d[base+3] = (__bf16)lo.w;
    d[base+4] = (__bf16)hi.x; d[base+5] = (__bf16)hi.y;
    d[base+6] = (__bf16)hi.z; d[base+7] = (__bf16)hi.w;
}

// Fused MLP encoder: Z = relu(x @ W1^T + b1) @ W2^T + b2
// One wave per 16-node tile; GEMM1 via v_wmma_f32_16x16x32_bf16, GEMM2 (K=64,
// 2 outputs) via f32 VALU through LDS.
__global__ __launch_bounds__(256) void encoder_wmma(
    const float* __restrict__ x,  const float* __restrict__ W1,
    const float* __restrict__ b1, const float* __restrict__ W2,
    const float* __restrict__ b2, float* __restrict__ Z, int N)
{
    __shared__ float hlds[8][16 * 65];   // per-wave 16x64 tile, stride 65 (bank-conflict pad)

    const int lane = threadIdx.x & 31;
    const int wave = threadIdx.x >> 5;
    const int half = lane >> 4;          // 0..1
    const int nl   = lane & 15;          // 0..15
    const int row0 = (blockIdx.x * 8 + wave) * 16;

    // A-matrix row for this lane (clamped; stores are guarded later)
    int arow = row0 + nl;
    if (arow >= N) arow = N - 1;
    const float* xrow = x + (size_t)arow * IN_DIM;

    v8f acc[4] = {};                     // 4 n-tiles of 16 hidden columns each

    #pragma unroll
    for (int c = 0; c < IN_DIM / 32; ++c) {
        const int kb = c * 32;
        // A fragment (16x32 bf16): K = kb + {8*half..8*half+7, 16+8*half..16+8*half+7}
        v16bf afrag;
        {
            const float4* p0 = (const float4*)(xrow + kb + half * 8);
            const float4* p1 = (const float4*)(xrow + kb + 16 + half * 8);
            pack8(afrag, 0, p0[0], p0[1]);
            pack8(afrag, 8, p1[0], p1[1]);
        }
        #pragma unroll
        for (int t = 0; t < 4; ++t) {
            // B fragment (32x16 bf16) = W1^T chunk: lane holds hidden col t*16+nl,
            // K = kb + {8*half..8*half+7, 16+8*half..16+8*half+7} (matches A)
            const float4* q = (const float4*)(W1 + (size_t)(t * 16 + nl) * IN_DIM
                                              + kb + half * 8);
            v16bf bfrag;
            pack8(bfrag, 0, q[0], q[1]);
            pack8(bfrag, 8, q[4], q[5]);
            acc[t] = __builtin_amdgcn_wmma_f32_16x16x32_bf16(
                false, afrag, false, bfrag, (short)0, acc[t], false, false);
        }
    }

    // bias + ReLU, spill C-layout tile to LDS: lane holds col n=nl, VGPR j -> row j+8*half
    float* hw = &hlds[wave][0];
    #pragma unroll
    for (int t = 0; t < 4; ++t) {
        const float bias = b1[t * 16 + nl];
        #pragma unroll
        for (int j = 0; j < 8; ++j) {
            float v = acc[t][j] + bias;
            v = v > 0.0f ? v : 0.0f;
            hw[(j + 8 * half) * 65 + t * 16 + nl] = v;
        }
    }
    __syncthreads();   // uniform: no early returns anywhere above

    // GEMM2: 16 rows x 2 outputs = 32 lanes, one (row,out) pair per lane
    const int mrow = lane >> 1;
    const int o    = lane & 1;
    float z = b2[o];
    const float* w2 = W2 + o * HIDDEN;
    #pragma unroll
    for (int k = 0; k < HIDDEN; ++k)
        z += hw[mrow * 65 + k] * w2[k];

    const int grow = row0 + mrow;
    if (grow < N) Z[(size_t)grow * ODIM + o] = z;
}

// H_next = alpha * Z   (teleport term; also serves as the accumulator init)
__global__ __launch_bounds__(256) void appnp_init(
    const float* __restrict__ Z, float* __restrict__ Hnext, int N)
{
    int i = blockIdx.x * blockDim.x + threadIdx.x;
    if (i < N) {
        float2 zv = ((const float2*)Z)[i];
        float2 r; r.x = ALPHA * zv.x; r.y = ALPHA * zv.y;
        ((float2*)Hnext)[i] = r;
    }
}

// H_next[row] += (1-alpha) * val * H_cur[col]   (gather + atomic scatter-add)
__global__ __launch_bounds__(256) void appnp_edge(
    const int* __restrict__ rows, const int* __restrict__ cols,
    const float* __restrict__ vals, const float* __restrict__ Hcur,
    float* __restrict__ Hnext, int E)
{
    int i = blockIdx.x * blockDim.x + threadIdx.x;
    if (i >= E) return;
    const int   r = rows[i];
    const int   c = cols[i];
    const float v = vals[i] * (1.0f - ALPHA);
    const float2 h = ((const float2*)Hcur)[c];
    atomicAdd(&Hnext[(size_t)r * ODIM + 0], v * h.x);
    atomicAdd(&Hnext[(size_t)r * ODIM + 1], v * h.y);
}

extern "C" void kernel_launch(void* const* d_in, const int* in_sizes, int n_in,
                              void* d_out, int out_size, void* d_ws, size_t ws_size,
                              hipStream_t stream) {
    const float* x    = (const float*)d_in[0];
    const float* W1   = (const float*)d_in[1];
    const float* b1   = (const float*)d_in[2];
    const float* W2   = (const float*)d_in[3];
    const float* b2   = (const float*)d_in[4];
    const int*   er   = (const int*)d_in[5];
    const int*   ec   = (const int*)d_in[6];
    const float* ev   = (const float*)d_in[7];

    const int N = in_sizes[0] / IN_DIM;   // 100000
    const int E = in_sizes[5];            // 3200000

    float* Zbuf = (float*)d_ws;           // [N*2]
    float* P0   = Zbuf + (size_t)N * ODIM;// [N*2] ping buffer
    float* P1   = (float*)d_out;          // pong buffer == final output (step 10 is even)

    // Encoder: 8 waves/block, 16 rows/wave -> 128 rows/block
    encoder_wmma<<<(N + 127) / 128, 256, 0, stream>>>(x, W1, b1, W2, b2, Zbuf, N);

    // APPNP power iteration, ping-pong so step K lands in d_out
    const float* cur = Zbuf;              // H0 = Z
    for (int step = 1; step <= KSTEPS; ++step) {
        float* nxt = (step & 1) ? P0 : P1;
        appnp_init<<<(N + 255) / 256, 256, 0, stream>>>(Zbuf, nxt, N);
        appnp_edge<<<(E + 255) / 256, 256, 0, stream>>>(er, ec, ev, cur, nxt, E);
        cur = nxt;
    }
}
